// SpatialOut_61503931678828
// MI455X (gfx1250) — matrix operand order repaired and checked by reading, not compiled, not run on hardware
//
#include <hip/hip_runtime.h>
#include <hip/hip_bf16.h>
#include <stdint.h>

// CDNA5 / gfx1250: wave32, WMMA 16x16x32 f16 -> f32
typedef __attribute__((ext_vector_type(16))) _Float16 v16h;
typedef __attribute__((ext_vector_type(8)))  float    v8f;

union BFrag { uint4 u4[2]; v16h h; };

#define NODE_DIM 128
#define HID 64
#define TILES_PER_BLOCK 4          // each wave handles 4 tiles of 16 atoms
#define ATOMS_PER_BLOCK (128 * TILES_PER_BLOCK)

__global__ __launch_bounds__(256, 2)
void spatialout_wmma_kernel(const float* __restrict__ xs,
                            const float* __restrict__ coord,
                            const long long* __restrict__ bidx,
                            const float* __restrict__ W1,
                            const float* __restrict__ b1,
                            const float* __restrict__ W2,
                            const float* __restrict__ b2,
                            float* __restrict__ out,
                            int nAtoms)
{
    // W1 (128x64 f32) converted to f16 and pre-shuffled into the per-lane
    // B-matrix fragment layout for v_wmma_f32_16x16x32_f16:
    //   frag(kk,n): lane l holds column N = n*16 + (l&15);
    //   VGPR v holds pair (K0 + (l>>4)*8 + (v>>2)*16 + 2*(v&3), +1), K0 = kk*32.
    __shared__ unsigned sB[4096];   // 16 KB
    __shared__ float sB1[HID];
    __shared__ float sW2[HID];

    const int tid = threadIdx.x;

    #pragma unroll
    for (int it = 0; it < 16; ++it) {
        int idx  = tid + it * 256;          // 0..4095
        int v    = idx & 7;
        int l    = (idx >> 3) & 31;
        int frag = idx >> 8;                // 0..15
        int n    = frag & 3;
        int kk   = frag >> 2;
        int k    = kk * 32 + (l >> 4) * 8 + (v >> 2) * 16 + 2 * (v & 3);
        int N    = n * 16 + (l & 15);
        _Float16 lo = (_Float16)W1[k * HID + N];
        _Float16 hi = (_Float16)W1[(k + 1) * HID + N];
        unsigned short ulo, uhi;
        __builtin_memcpy(&ulo, &lo, 2);
        __builtin_memcpy(&uhi, &hi, 2);
        sB[idx] = (unsigned)ulo | ((unsigned)uhi << 16);
    }
    if (tid < HID) { sB1[tid] = b1[tid]; sW2[tid] = W2[tid]; }
    __syncthreads();

    const int lane = tid & 31;       // wave32 lane
    const int wv   = tid >> 5;       // wave in block: 0..7
    const int hh   = lane >> 4;      // half-wave
    const int mm   = lane & 15;
    const float b2v = b2[0];

    for (int t = 0; t < TILES_PER_BLOCK; ++t) {
        const long tileBase = (long)blockIdx.x * ATOMS_PER_BLOCK
                            + (long)t * 128 + (long)wv * 16;
        long atomA = tileBase + mm;
        if (atomA >= nAtoms) atomA = nAtoms - 1;   // clamp (no divergence at WMMA)
        const float* row = xs + atomA * NODE_DIM;

        // Keep the HBM stream ahead: prefetch this wave's next-tile row
        // (clamped so the address is always mapped). -> global_prefetch_b8
        long atomP = tileBase + 128 + mm;
        if (atomP >= nAtoms) atomP = nAtoms - 1;
        const float* rowP = xs + atomP * NODE_DIM;
        #pragma unroll
        for (int kk = 0; kk < 4; ++kk)
            __builtin_prefetch(rowP + kk * 32, 0, 0);

        // Don't let the compiler hoist the LDS B-fragment reads out of the
        // tile loop (that costs ~128 VGPRs and strangles the load pipeline).
        asm volatile("" ::: "memory");

        // Issue ALL 16 A-tile loads before any conversion/wait so they form
        // one clause with 16 loads in flight.
        float4 a[16];
        #pragma unroll
        for (int kk = 0; kk < 4; ++kk) {
            const int kb = kk * 32 + hh * 8;
            a[kk * 4 + 0] = *reinterpret_cast<const float4*>(row + kb);
            a[kk * 4 + 1] = *reinterpret_cast<const float4*>(row + kb + 4);
            a[kk * 4 + 2] = *reinterpret_cast<const float4*>(row + kb + 16);
            a[kk * 4 + 3] = *reinterpret_cast<const float4*>(row + kb + 20);
        }

        // f32 -> f16 A fragments (A layout: half h holds K0+h*8+{0..7} then K0+16+h*8+{0..7})
        v16h Af[4];
        #pragma unroll
        for (int kk = 0; kk < 4; ++kk) {
            float4 a0 = a[kk * 4 + 0], a1 = a[kk * 4 + 1];
            float4 a2 = a[kk * 4 + 2], a3 = a[kk * 4 + 3];
            v16h A;
            A[0]=(_Float16)a0.x;  A[1]=(_Float16)a0.y;  A[2]=(_Float16)a0.z;  A[3]=(_Float16)a0.w;
            A[4]=(_Float16)a1.x;  A[5]=(_Float16)a1.y;  A[6]=(_Float16)a1.z;  A[7]=(_Float16)a1.w;
            A[8]=(_Float16)a2.x;  A[9]=(_Float16)a2.y;  A[10]=(_Float16)a2.z; A[11]=(_Float16)a2.w;
            A[12]=(_Float16)a3.x; A[13]=(_Float16)a3.y; A[14]=(_Float16)a3.z; A[15]=(_Float16)a3.w;
            Af[kk] = A;
        }

        v8f acc[4] = {};   // 4 N-tiles of the 16x64 hidden output
        #pragma unroll
        for (int kk = 0; kk < 4; ++kk) {
            #pragma unroll
            for (int n = 0; n < 4; ++n) {
                BFrag B;
                const uint4* p = reinterpret_cast<const uint4*>(&sB[(kk * 4 + n) * 256 + lane * 8]);
                B.u4[0] = p[0];
                B.u4[1] = p[1];
                acc[n] = __builtin_amdgcn_wmma_f32_16x16x32_f16(
                             false, Af[kk], false, B.h, (short)0, acc[n], false, false);
            }
        }

        // bias + SiLU + dot with W2, reduced over the N (hidden) dimension.
        // C layout: lane l holds N = l&15 for rows M = 8*(l>>4) + r, r = vgpr 0..7.
        float part[8] = {0.f,0.f,0.f,0.f,0.f,0.f,0.f,0.f};
        #pragma unroll
        for (int n = 0; n < 4; ++n) {
            int j = n * 16 + mm;
            float b1j = sB1[j];
            float w2j = sW2[j];
            #pragma unroll
            for (int r = 0; r < 8; ++r) {
                float av = acc[n][r] + b1j;
                float s = av / (1.0f + __expf(-av));   // SiLU
                part[r] += s * w2j;
            }
        }
        // Reduce across the 16 lanes of each half-wave (masks < 16 keep halves separate).
        #pragma unroll
        for (int r = 0; r < 8; ++r) {
            #pragma unroll
            for (int m = 1; m < 16; m <<= 1)
                part[r] += __shfl_xor(part[r], m, 32);
        }

        // One writer lane per atom row: lanes with (lane&15) < 8 handle M = 8*hh + (lane&7).
        if (mm < 8) {
            long atom = tileBase + hh * 8 + mm;
            if (atom < nAtoms) {
                float so = part[mm] + b2v;
                const float* cp = coord + atom * 3;
                float cx = cp[0], cy = cp[1], cz = cp[2];
                float sp = cx * cx + cy * cy + cz * cz;
                long mol = (long)bidx[atom];
                unsafeAtomicAdd(&out[mol], so * sp);   // global_atomic_add_f32
            }
        }
    }
}

extern "C" void kernel_launch(void* const* d_in, const int* in_sizes, int n_in,
                              void* d_out, int out_size, void* d_ws, size_t ws_size,
                              hipStream_t stream) {
    const float*     xs    = (const float*)d_in[0];
    // d_in[1] = x_spherical (unused by the reference)
    const float*     coord = (const float*)d_in[2];
    const long long* bidx  = (const long long*)d_in[3];   // int64 batch index
    const float*     W1    = (const float*)d_in[4];
    const float*     b1    = (const float*)d_in[5];
    const float*     W2    = (const float*)d_in[6];
    const float*     b2    = (const float*)d_in[7];
    float*           out   = (float*)d_out;

    const int nAtoms = in_sizes[0] / NODE_DIM;

    // We accumulate with atomics, so the output must be zeroed every call
    // (memset nodes are graph-capturable).
    hipMemsetAsync(d_out, 0, (size_t)out_size * sizeof(float), stream);

    const int blocks = (nAtoms + ATOMS_PER_BLOCK - 1) / ATOMS_PER_BLOCK;
    spatialout_wmma_kernel<<<blocks, 256, 0, stream>>>(
        xs, coord, bidx, W1, b1, W2, b2, out, nAtoms);
}
